// SwinBlock_8632884265097
// MI455X (gfx1250) — matrix operand order repaired and checked by reading, not compiled, not run on hardware
//
#include <hip/hip_runtime.h>
#include <hip/hip_bf16.h>
#include <math.h>

// ---------------- types for WMMA ----------------
typedef __attribute__((ext_vector_type(16))) _Float16 v16h;
typedef __attribute__((ext_vector_type(2)))  _Float16 v2h;
typedef __attribute__((ext_vector_type(8)))  float    v8f;
typedef __attribute__((ext_vector_type(4)))  int      vint4;

union Frag16 { v16h v; v2h p[8]; };

// ---------------- async-to-LDS plumbing (gfx1250) ----------------
#if defined(__HIP_DEVICE_COMPILE__) && __has_builtin(__builtin_amdgcn_global_load_async_to_lds_b128)
#define GEMM_ASYNC 1
typedef __attribute__((address_space(1))) vint4 g_vint4;
typedef __attribute__((address_space(3))) vint4 l_vint4;
#define WAIT_ASYNC(n) asm volatile("s_wait_asynccnt " #n ::: "memory")
#else
#define GEMM_ASYNC 0
#define WAIT_ASYNC(n) ((void)0)
#endif

// ---------------- problem constants ----------------
#define BB     16
#define HH     56
#define CC     192
#define LTOK   50176          // 16*56*56
#define NHEAD  6
#define HDIM   32
#define WS     7
#define SHIFT  3
#define NTOK   49             // tokens per window
#define NWIN   1024           // 16 * 8 * 8
#define QKVN   576
#define HIDN   1152

// window-row -> token index (applies the +SHIFT unroll; same map both ways)
__device__ __forceinline__ int win2tok(int r) {
    int wb = r / NTOK, n = r % NTOK;
    int b  = wb >> 6, wrem = wb & 63;
    int wi = wrem >> 3, wj = wrem & 7;
    int ni = n / WS,  nj = n % WS;
    int hh = wi * WS + ni + SHIFT; if (hh >= HH) hh -= HH;
    int ww = wj * WS + nj + SHIFT; if (ww >= HH) ww -= HH;
    return (b * HH + hh) * HH + ww;
}

// ---------------- f32 -> f16 weight convert ----------------
__global__ void cvt_f16_kernel(const float* __restrict__ src,
                               _Float16* __restrict__ dst, int n) {
    int i = blockIdx.x * blockDim.x + threadIdx.x;
    if (i < n) dst[i] = (_Float16)src[i];
}

// ---------------- precompute attention bias table [head][64][64] ----------------
// invalid (padded) entries hold the -1e30 softmax sentinel
__global__ void bias_tab_kernel(const float* __restrict__ rpb,
                                float* __restrict__ tab) {
    int i = blockIdx.x * 256 + threadIdx.x;          // 6*64*64 = 24576
    int head = i >> 12, row = (i >> 6) & 63, col = i & 63;
    float v = -1e30f;
    if (row < NTOK && col < NTOK) {
        int idx = (row / WS - col / WS + WS - 1) * (2 * WS - 1)
                + (row % WS - col % WS + WS - 1);
        v = rpb[idx * NHEAD + head];
    }
    tab[i] = v;
}

// ---------------- LayerNorm (optionally fused with window gather) ----------------
template <bool GATHER>
__global__ __launch_bounds__(256) void ln_kernel(const float* __restrict__ x,
                                                 const float* __restrict__ g,
                                                 const float* __restrict__ b,
                                                 _Float16* __restrict__ out) {
    int wave = threadIdx.x >> 5, lane = threadIdx.x & 31;
    int r = blockIdx.x * 8 + wave;                 // 50176 / 8 = 6272 blocks, exact
    int src = GATHER ? win2tok(r) : r;
    const float* xr = x + (size_t)src * CC;
    float v[6];
    float s = 0.f;
    #pragma unroll
    for (int i = 0; i < 6; ++i) { v[i] = xr[lane + 32 * i]; s += v[i]; }
    #pragma unroll
    for (int m = 16; m > 0; m >>= 1) s += __shfl_xor(s, m, 32);
    float mean = s * (1.f / CC);
    float vs = 0.f;
    #pragma unroll
    for (int i = 0; i < 6; ++i) { float d = v[i] - mean; vs += d * d; }
    #pragma unroll
    for (int m = 16; m > 0; m >>= 1) vs += __shfl_xor(vs, m, 32);
    float rstd = rsqrtf(vs * (1.f / CC) + 1e-5f);
    _Float16* orow = out + (size_t)r * CC;
    #pragma unroll
    for (int i = 0; i < 6; ++i) {
        int c = lane + 32 * i;
        orow[c] = (_Float16)((v[i] - mean) * rstd * g[c] + b[c]);
    }
}

// ---------------- WMMA GEMM, 64x64 block tile, double-buffered async LDS ----------
// out = A(MxK) * W(NxK)^T + bias
// EPI: 0 = store f16; 1 = exact GELU -> f16; 2 = proj: window->token map,
//      +resid(x) -> f32 (token order); 3 = +resid -> f32 (same order)
template <int EPI>
__global__ __launch_bounds__(256) void gemm_wmma_kernel(
        const _Float16* __restrict__ A, const _Float16* __restrict__ W,
        const float* __restrict__ bias,
        _Float16* __restrict__ outH, float* __restrict__ outF,
        const float* __restrict__ resid,
        int M, int N, int K) {
    __shared__ _Float16 sA[2][64 * 32];
    __shared__ _Float16 sB[2][64 * 32];

    const int tid  = threadIdx.x;
    const int wave = tid >> 5, lane = tid & 31, half = lane >> 4;
    const int m0 = blockIdx.y * 64;
    const int n0 = blockIdx.x * 64;
    const int nst = K >> 5;

    // each thread moves one 16B chunk of A and one of B per k-stage
    const int r   = tid >> 2;
    const int seg = (tid & 3) << 3;
    const _Float16* gA = A + (size_t)(m0 + r) * K + seg;
    const _Float16* gB = W + (size_t)(n0 + r) * K + seg;
    const int ldso = r * 32 + seg;

    auto issue = [&](int s, int buf) {
#if GEMM_ASYNC
        __builtin_amdgcn_global_load_async_to_lds_b128(
            (g_vint4*)(gA + s * 32), (l_vint4*)&sA[buf][ldso], 0, 0);
        __builtin_amdgcn_global_load_async_to_lds_b128(
            (g_vint4*)(gB + s * 32), (l_vint4*)&sB[buf][ldso], 0, 0);
#else
        *(vint4*)&sA[buf][ldso] = *(const vint4*)(gA + s * 32);
        *(vint4*)&sB[buf][ldso] = *(const vint4*)(gB + s * 32);
#endif
    };

    // wave -> one 16-row tile, two 16-col tiles
    const int mt   = wave >> 1;
    const int ntp  = (wave & 1) << 1;
    const int arow = mt * 16 + (lane & 15);
    const int bn0  = ntp * 16 + (lane & 15);
    const int bn1  = bn0 + 16;

    v8f acc0 = {}, acc1 = {};
    issue(0, 0);
    for (int s = 0; s < nst; ++s) {
        const int cur = s & 1;
        if (s + 2 < nst) {
            __builtin_prefetch(gA + (s + 2) * 32, 0, 0);
            __builtin_prefetch(gB + (s + 2) * 32, 0, 0);
        }
        if (s + 1 < nst) {
            issue(s + 1, cur ^ 1);
            WAIT_ASYNC(2);
        } else {
            WAIT_ASYNC(0);
        }
        __syncthreads();
        Frag16 a, b0, b1;
        #pragma unroll
        for (int j = 0; j < 8; ++j) {
            int ka = (half << 4) + (j << 1);
            a.p[j]  = *(const v2h*)&sA[cur][arow * 32 + ka];
            int kb = (half << 4) + (j << 1);
            b0.p[j] = *(const v2h*)&sB[cur][bn0 * 32 + kb];
            b1.p[j] = *(const v2h*)&sB[cur][bn1 * 32 + kb];
        }
        acc0 = __builtin_amdgcn_wmma_f32_16x16x32_f16(
                false, a.v, false, b0.v, (short)0, acc0, false, false);
        acc1 = __builtin_amdgcn_wmma_f32_16x16x32_f16(
                false, a.v, false, b1.v, (short)0, acc1, false, false);
        __syncthreads();
    }

    const int rbase = m0 + mt * 16 + (half << 3);
    auto writeTile = [&](const v8f& c, int nt) {
        int col = n0 + nt * 16 + (lane & 15);
        float bb = bias[col];
        #pragma unroll
        for (int j = 0; j < 8; ++j) {
            int row = rbase + j;
            float val = c[j] + bb;
            if (EPI == 0) {
                outH[(size_t)row * N + col] = (_Float16)val;
            } else if (EPI == 1) {
                val = 0.5f * val * (1.f + erff(val * 0.70710678118654752f));
                outH[(size_t)row * N + col] = (_Float16)val;
            } else if (EPI == 2) {
                size_t o = (size_t)win2tok(row) * N + col;
                outF[o] = val + resid[o];
            } else {
                size_t o = (size_t)row * N + col;
                outF[o] = val + resid[o];
            }
        }
    };
    writeTile(acc0, ntp);
    writeTile(acc1, ntp + 1);
}

// ---------------- per-(window, head) attention ----------------
__global__ __launch_bounds__(128) void attn_kernel(const _Float16* __restrict__ qkv,
                                                   const float* __restrict__ biasTab,
                                                   _Float16* __restrict__ attn_out) {
    __shared__ _Float16 sQ[64 * 32];
    __shared__ _Float16 sK[64 * 32];
    __shared__ _Float16 sVt[32 * 64];    // V transposed: [d][token]
    __shared__ _Float16 sP[64 * 64];
    __shared__ float    sS[64 * 64];

    int wb = blockIdx.x, head = blockIdx.y;
    int tid = threadIdx.x;
    const float scale = 0.17677669529663687f;   // 32^-0.5
    size_t base = (size_t)wb * NTOK * QKVN + head * HDIM;
    const float* bTab = biasTab + (size_t)head * 4096;

    for (int i = tid; i < 64 * 32; i += 128) {
        int r = i >> 5, d = i & 31;
        float qv = 0.f, kv = 0.f, vv = 0.f;
        if (r < NTOK) {
            size_t o = base + (size_t)r * QKVN + d;
            qv = (float)qkv[o] * scale;
            kv = (float)qkv[o + CC];
            vv = (float)qkv[o + 2 * CC];
        }
        sQ[i] = (_Float16)qv;
        sK[i] = (_Float16)kv;
        sVt[d * 64 + r] = (_Float16)vv;
    }
    __syncthreads();

    int wave = tid >> 5, lane = tid & 31, half = lane >> 4;
    int mt = wave;                       // 4 waves -> 4 row tiles of S

    // ---- S = Q K^T  (K=32 -> one WMMA step), + precomputed bias/sentinel ----
    Frag16 aq;
    {
        int arow = mt * 16 + (lane & 15);
        #pragma unroll
        for (int j = 0; j < 8; ++j) {
            int ka = (half << 4) + (j << 1);
            aq.p[j] = *(const v2h*)&sQ[arow * 32 + ka];
        }
    }
    #pragma unroll
    for (int nt = 0; nt < 4; ++nt) {
        Frag16 bk;
        int bn = nt * 16 + (lane & 15);
        #pragma unroll
        for (int j = 0; j < 8; ++j) {
            int kb = (half << 4) + (j << 1);
            bk.p[j] = *(const v2h*)&sK[bn * 32 + kb];
        }
        v8f c = {};
        c = __builtin_amdgcn_wmma_f32_16x16x32_f16(
                false, aq.v, false, bk.v, (short)0, c, false, false);
        int col = nt * 16 + (lane & 15);
        int rbase = mt * 16 + (half << 3);
        #pragma unroll
        for (int j = 0; j < 8; ++j) {
            int row = rbase + j;
            sS[row * 64 + col] = c[j] + bTab[row * 64 + col];
        }
    }
    __syncthreads();

    // ---- softmax over rows (sentinels underflow to 0 in exp) ----
    if (tid < 64) {
        int row = tid;
        if (row < NTOK) {
            float mx = -1e30f;
            for (int c = 0; c < 64; ++c) mx = fmaxf(mx, sS[row * 64 + c]);
            float sum = 0.f;
            for (int c = 0; c < 64; ++c) {
                float e = __expf(sS[row * 64 + c] - mx);
                sS[row * 64 + c] = e;
                sum += e;
            }
            float inv = 1.f / sum;
            for (int c = 0; c < 64; ++c)
                sP[row * 64 + c] = (_Float16)(sS[row * 64 + c] * inv);
        } else {
            for (int c = 0; c < 64; ++c) sP[row * 64 + c] = (_Float16)0.f;
        }
    }
    __syncthreads();

    // ---- O = P V  (K=64 -> two WMMA steps) ----
    #pragma unroll
    for (int nt = 0; nt < 2; ++nt) {
        v8f c = {};
        #pragma unroll
        for (int kk = 0; kk < 64; kk += 32) {
            Frag16 ap, bv;
            int arow = mt * 16 + (lane & 15);
            int bn = nt * 16 + (lane & 15);
            #pragma unroll
            for (int j = 0; j < 8; ++j) {
                int ka = kk + (half << 4) + (j << 1);
                ap.p[j] = *(const v2h*)&sP[arow * 64 + ka];
                int kb = kk + (half << 4) + (j << 1);
                bv.p[j] = *(const v2h*)&sVt[bn * 64 + kb];
            }
            c = __builtin_amdgcn_wmma_f32_16x16x32_f16(
                    false, ap.v, false, bv.v, (short)0, c, false, false);
        }
        int col = nt * 16 + (lane & 15);
        int rbase = mt * 16 + (half << 3);
        #pragma unroll
        for (int j = 0; j < 8; ++j) {
            int row = rbase + j;
            if (row < NTOK)
                attn_out[((size_t)wb * NTOK + row) * CC + head * HDIM + col] = (_Float16)c[j];
        }
    }
}

// ---------------- launcher ----------------
extern "C" void kernel_launch(void* const* d_in, const int* in_sizes, int n_in,
                              void* d_out, int out_size, void* d_ws, size_t ws_size,
                              hipStream_t stream) {
    const float* x      = (const float*)d_in[0];
    const float* ln1_g  = (const float*)d_in[3];
    const float* ln1_b  = (const float*)d_in[4];
    const float* qkv_w  = (const float*)d_in[5];
    const float* qkv_b  = (const float*)d_in[6];
    const float* rpb    = (const float*)d_in[7];
    const float* proj_w = (const float*)d_in[8];
    const float* proj_b = (const float*)d_in[9];
    const float* ln2_g  = (const float*)d_in[10];
    const float* ln2_b  = (const float*)d_in[11];
    const float* fc1_w  = (const float*)d_in[12];
    const float* fc1_b  = (const float*)d_in[13];
    const float* fc2_w  = (const float*)d_in[14];
    const float* fc2_b  = (const float*)d_in[15];
    float* out = (float*)d_out;

    char* ws = (char*)d_ws;
    auto take = [&](size_t bytes) { char* r = ws; ws += bytes; return r; };
    _Float16* w16_qkv  = (_Float16*)take((size_t)QKVN * CC * 2);
    _Float16* w16_proj = (_Float16*)take((size_t)CC * CC * 2);
    _Float16* w16_fc1  = (_Float16*)take((size_t)HIDN * CC * 2);
    _Float16* w16_fc2  = (_Float16*)take((size_t)CC * HIDN * 2);
    float*    biasTab  = (float*)   take((size_t)NHEAD * 64 * 64 * 4);
    _Float16* xn_win   = (_Float16*)take((size_t)LTOK * CC * 2);
    _Float16* qkv_h    = (_Float16*)take((size_t)LTOK * QKVN * 2);
    _Float16* attn_h   = (_Float16*)take((size_t)LTOK * CC * 2);
    float*    x1       = (float*)   take((size_t)LTOK * CC * 4);
    _Float16* h2n      = (_Float16*)take((size_t)LTOK * CC * 2);
    _Float16* fc1h     = (_Float16*)take((size_t)LTOK * HIDN * 2);

    // 1) weights -> f16, bias table
    cvt_f16_kernel<<<(QKVN * CC + 255) / 256, 256, 0, stream>>>(qkv_w, w16_qkv, QKVN * CC);
    cvt_f16_kernel<<<(CC * CC + 255) / 256, 256, 0, stream>>>(proj_w, w16_proj, CC * CC);
    cvt_f16_kernel<<<(HIDN * CC + 255) / 256, 256, 0, stream>>>(fc1_w, w16_fc1, HIDN * CC);
    cvt_f16_kernel<<<(CC * HIDN + 255) / 256, 256, 0, stream>>>(fc2_w, w16_fc2, CC * HIDN);
    bias_tab_kernel<<<(NHEAD * 64 * 64) / 256, 256, 0, stream>>>(rpb, biasTab);

    // 2) LN1 + shifted-window gather (token order -> window order, f16)
    ln_kernel<true><<<LTOK / 8, 256, 0, stream>>>(x, ln1_g, ln1_b, xn_win);

    // 3) QKV projection: (50176 x 192) @ (576 x 192)^T
    gemm_wmma_kernel<0><<<dim3(QKVN / 64, LTOK / 64), 256, 0, stream>>>(
        xn_win, w16_qkv, qkv_b, qkv_h, nullptr, nullptr, LTOK, QKVN, CC);

    // 4) windowed attention (per window, per head)
    attn_kernel<<<dim3(NWIN, NHEAD), 128, 0, stream>>>(qkv_h, biasTab, attn_h);

    // 5) proj + window-reverse/shift + residual -> x1 (token order, f32)
    gemm_wmma_kernel<2><<<dim3(CC / 64, LTOK / 64), 256, 0, stream>>>(
        attn_h, w16_proj, proj_b, nullptr, x1, x, LTOK, CC, CC);

    // 6) LN2
    ln_kernel<false><<<LTOK / 8, 256, 0, stream>>>(x1, ln2_g, ln2_b, h2n);

    // 7) FC1 + exact GELU
    gemm_wmma_kernel<1><<<dim3(HIDN / 64, LTOK / 64), 256, 0, stream>>>(
        h2n, w16_fc1, fc1_b, fc1h, nullptr, nullptr, LTOK, HIDN, CC);

    // 8) FC2 + residual -> out
    gemm_wmma_kernel<3><<<dim3(CC / 64, LTOK / 64), 256, 0, stream>>>(
        fc1h, w16_fc2, fc2_b, nullptr, out, x1, LTOK, CC, HIDN);
}
